// DistillLossSimpleMSE_28896539967627
// MI455X (gfx1250) — matrix operand order repaired and checked by reading, not compiled, not run on hardware
//
#include <hip/hip_runtime.h>
#include <hip/hip_bf16.h>

// Problem constants (from reference): N_OBJ=8, P=65536, M=32, D=128
#define O_      8
#define P_      65536
#define M_      32
#define D_      128
#define DE      144        // extended cols: 0..127=x, 128=xsq, 129=1.0, 130..143=0
#define CHUNK   2048       // points per workgroup
#define KSTEP   32         // points per WMMA K-step (bf16 K=32)
#define NSTEPS  (CHUNK / KSTEP)

#define XH_STR  34         // halfword stride of bf16 planes (= 17 dwords, odd -> few conflicts)
#define XH_STRW (XH_STR / 2)
#define MS_STRD 18         // dword stride of packed mask pairs (conflict-checked)

#if defined(__has_builtin)
#  if __has_builtin(__builtin_amdgcn_global_load_async_to_lds_b128)
#    define HAS_ASYNC 1
#  endif
#endif

typedef __attribute__((ext_vector_type(16))) __bf16        v16bf;
typedef __attribute__((ext_vector_type(8)))  float         v8f;
typedef __attribute__((ext_vector_type(8)))  unsigned int  v8u;

static __device__ __forceinline__ v8f wmma_bf16(v16bf a, v16bf b, v8f c) {
    return __builtin_amdgcn_wmma_f32_16x16x32_bf16(
        /*neg_a=*/false, a, /*neg_b=*/false, b,
        /*c_mod=*/(short)0, c, /*reuse_a=*/false, /*reuse_b=*/false);
}

// 16-byte global -> LDS copy: CDNA5 async DMA path when available.
// Builtin signature (from hipcc diagnostic): param1 = as(1) int4*, so:
//   (global int4*, lds int4*, imm offset, imm cpol)
static __device__ __forceinline__ void cp16(const void* g, void* l) {
#if defined(HAS_ASYNC)
    typedef int v4i_ __attribute__((__vector_size__(4 * sizeof(int))));
    typedef __attribute__((address_space(1))) v4i_* gptr_t;
    typedef __attribute__((address_space(3))) v4i_* lptr_t;
    __builtin_amdgcn_global_load_async_to_lds_b128(
        (gptr_t)(void*)(g), (lptr_t)(l), /*offset=*/0, /*cpol=*/0);
#else
    *(float4*)l = *(const float4*)g;
#endif
}

static __device__ __forceinline__ void await_async() {
#if defined(HAS_ASYNC)
    asm volatile("s_wait_asynccnt 0x0" ::: "memory");
#endif
}

// split fp32 into bf16 hi/lo and store to the two planes
static __device__ __forceinline__ void put_split(__bf16* hp, __bf16* lp, float v) {
    __bf16 h = (__bf16)v;
    *hp = h;
    *lp = (__bf16)(v - (float)h);
}

__global__ __launch_bounds__(256) void zero_acc_kernel(float* __restrict__ acc, int n) {
    int i = blockIdx.x * blockDim.x + threadIdx.x;
    if (i < n) acc[i] = 0.0f;
}

// acc[o][m][c]: c<128 -> sum_p m*x[p][c]; c==128 -> sum_p m*xsq(p); c==129 -> cnt
__global__ __launch_bounds__(256) void partial_kernel(
    const float* __restrict__ net_out,    // [O*P][D]
    const int*   __restrict__ mask_pts,   // [O][M][P]
    float*       __restrict__ acc)        // [O][M][DE]
{
    __shared__ float rawx[2][KSTEP * D_];             // double-buffered raw x tile (2x16KB)
    __shared__ int   rawm[2][M_ * KSTEP];             // double-buffered raw mask tile (2x4KB)
    __shared__ __attribute__((aligned(16))) __bf16 xhb[DE][XH_STR];   // bf16-hi plane [col][pt]
    __shared__ __attribute__((aligned(16))) __bf16 xlb[DE][XH_STR];   // bf16-lo plane [col][pt]
    __shared__ unsigned int ms16[M_][MS_STRD];        // packed bf16 mask pairs [m][pt/2]
    __shared__ float xsqp[KSTEP][8];                  // xsq partials

    const int tid  = threadIdx.x;
    const int wave = tid >> 5;
    const int lane = tid & 31;

    const int obj   = blockIdx.x / (P_ / CHUNK);
    const int chunk = blockIdx.x % (P_ / CHUNK);
    const long prow0 = (long)obj * P_ + (long)chunk * CHUNK;   // first global x row
    const int  ploc0 = chunk * CHUNK;                          // first point-in-object

    // 18 output tiles = 2 mtiles x 9 dtiles; wave w owns tiles {w, w+8, w+16}
    const int  mtile = wave & 1;
    const int  dt0   = wave >> 1;
    const int  dt1   = dt0 + 4;
    const int  dt2   = dt0 + 8;              // valid only for waves 0,1
    const bool has3  = (dt2 < 9);

    v8f a0 = {}, a1 = {}, a2 = {};

    // One-time constant extended cols: 129 = 1.0 (hi=0x3F80, lo=0), 130..143 = 0
    for (int i = tid; i < 15 * KSTEP; i += 256) {
        int c = 129 + (i >> 5), p = i & 31;
        unsigned short hv = (c == 129) ? (unsigned short)0x3F80 : (unsigned short)0;
        xhb[c][p] = __builtin_bit_cast(__bf16, hv);
        xlb[c][p] = __builtin_bit_cast(__bf16, (unsigned short)0);
    }

    // ---- prologue: issue copies for step 0 into buffer 0 ----
    {
        const char* gx = (const char*)(net_out + prow0 * (long)D_);
        char*       lx = (char*)&rawx[0][0];
        #pragma unroll
        for (int q = 0; q < 4; ++q)
            cp16(gx + tid * 64 + q * 16, lx + tid * 64 + q * 16);
        const int m = tid >> 3, qq = tid & 7;
        const char* gm = (const char*)(mask_pts + ((long)obj * M_ + m) * P_ + ploc0);
        cp16(gm + qq * 16, (char*)&rawm[0][0] + m * 128 + qq * 16);
    }

    for (int kk = 0; kk < NSTEPS; ++kk) {
        const int cur = kk & 1, nxt = cur ^ 1;

        await_async();        // this wave's raw[cur] pieces have landed
        __syncthreads();      // all waves' pieces visible; prev step's planes consumed

        // ---- issue async copies for step kk+1 into raw[nxt] ----
        if (kk + 1 < NSTEPS) {
            const long prow = prow0 + (long)(kk + 1) * KSTEP;
            const char* gx = (const char*)(net_out + prow * (long)D_);
            char*       lx = (char*)&rawx[nxt][0];
            #pragma unroll
            for (int q = 0; q < 4; ++q)
                cp16(gx + tid * 64 + q * 16, lx + tid * 64 + q * 16);
            const int m = tid >> 3, qq = tid & 7;
            const char* gm = (const char*)(mask_pts +
                ((long)obj * M_ + m) * P_ + (ploc0 + (kk + 1) * KSTEP));
            cp16(gm + qq * 16, (char*)&rawm[nxt][0] + m * 128 + qq * 16);
        }

        // ---- convert x: fp32 -> bf16 hi/lo planes (col-major, point pairs adjacent) ----
        {
            const int p = tid >> 3, seg = tid & 7;
            const float4* src = (const float4*)&rawx[cur][p * D_ + seg * 16];
            float s = 0.0f;
            #pragma unroll
            for (int q = 0; q < 4; ++q) {
                float4 v = src[q];
                s += v.x * v.x + v.y * v.y + v.z * v.z + v.w * v.w;
                const int c = seg * 16 + q * 4;
                put_split(&xhb[c + 0][p], &xlb[c + 0][p], v.x);
                put_split(&xhb[c + 1][p], &xlb[c + 1][p], v.y);
                put_split(&xhb[c + 2][p], &xlb[c + 2][p], v.z);
                put_split(&xhb[c + 3][p], &xlb[c + 3][p], v.w);
            }
            xsqp[p][seg] = s;
        }
        // ---- convert mask: int 0/1 -> packed bf16 pairs (A-layout-ready dwords) ----
        {
            const int m = tid >> 3, qq = tid & 7;
            const int4 v = *(const int4*)&rawm[cur][m * KSTEP + qq * 4];
            ms16[m][qq * 2 + 0] = (v.x ? 0x3F80u : 0u) | ((v.y ? 0x3F80u : 0u) << 16);
            ms16[m][qq * 2 + 1] = (v.z ? 0x3F80u : 0u) | ((v.w ? 0x3F80u : 0u) << 16);
        }
        __syncthreads();

        // ---- finish extended col 128 (xsq) ----
        if (tid < KSTEP) {
            float s = 0.0f;
            #pragma unroll
            for (int q = 0; q < 8; ++q) s += xsqp[tid][q];
            put_split(&xhb[128][tid], &xlb[128][tid], s);
        }
        __syncthreads();

        // ---- gather fragments (pure dword LDS loads, zero conversion VALU) ----
        const unsigned int* xhw = (const unsigned int*)&xhb[0][0];
        const unsigned int* xlw = (const unsigned int*)&xlb[0][0];
        const unsigned int* msw = &ms16[0][0];
        const int cl  = lane & 15;
        const int kb2 = (lane >> 4) << 3;     // K offset in point-pairs (B layout)
        const int kh2 = (lane >> 4) << 3;     // K offset in point-pairs (A layout)

        v8u au;
        {
            const int mrow = mtile * 16 + cl;
            #pragma unroll
            for (int v = 0; v < 8; ++v)       // A 16x32 bf16 layout, pair-packed
                au[v] = msw[mrow * MS_STRD + kh2 + v];
        }
        const v16bf af = __builtin_bit_cast(v16bf, au);

        auto do_tile = [&](int dt, v8f& acc8) {
            const int base = (dt * 16 + cl) * XH_STRW + kb2;   // B 32x16 bf16 layout
            v8u bh, bl;
            #pragma unroll
            for (int v = 0; v < 8; ++v) { bh[v] = xhw[base + v]; bl[v] = xlw[base + v]; }
            acc8 = wmma_bf16(af, __builtin_bit_cast(v16bf, bh), acc8);
            acc8 = wmma_bf16(af, __builtin_bit_cast(v16bf, bl), acc8);
        };
        do_tile(dt0, a0);
        do_tile(dt1, a1);
        if (has3) do_tile(dt2, a2);   // wave-uniform branch: EXEC all-ones for WMMA
    }

    // ---- write back accumulators (C layout: row M = v + 8*(lane>>4), col N = lane&15) ----
    const int mrow = mtile * 16 + ((lane >> 4) << 3);
    const int ncol = lane & 15;
    #pragma unroll
    for (int v = 0; v < 8; ++v) {
        atomicAdd(&acc[((long)obj * M_ + (mrow + v)) * DE + dt0 * 16 + ncol], a0[v]);
        atomicAdd(&acc[((long)obj * M_ + (mrow + v)) * DE + dt1 * 16 + ncol], a1[v]);
    }
    if (has3) {
        #pragma unroll
        for (int v = 0; v < 8; ++v)
            atomicAdd(&acc[((long)obj * M_ + (mrow + v)) * DE + dt2 * 16 + ncol], a2[v]);
    }
}

__global__ __launch_bounds__(256) void finalize_kernel(
    const float* __restrict__ acc,        // [O*M][DE]
    const float* __restrict__ mask_embs,  // [O*M][D]
    float*       __restrict__ out)
{
    __shared__ float red_s[256];
    __shared__ float red_c[256];
    const int i = threadIdx.x;            // one (o,m) pair per thread, 256 total

    const float* row = acc + (long)i * DE;
    const float* e   = mask_embs + (long)i * D_;
    float t1  = row[128];
    float cnt = row[129];
    float esq = 0.0f, dot = 0.0f;
    #pragma unroll 4
    for (int d = 0; d < D_; ++d) {
        float ev = e[d];
        if (!isfinite(ev)) ev = 0.0f;     // nan_to_num(nan=0, posinf=0, neginf=0)
        esq += ev * ev;
        dot += ev * row[d];
    }
    red_s[i] = t1 + cnt * esq - 2.0f * dot;
    red_c[i] = cnt;
    __syncthreads();
    #pragma unroll
    for (int off = 128; off > 0; off >>= 1) {
        if (i < off) { red_s[i] += red_s[i + off]; red_c[i] += red_c[i + off]; }
        __syncthreads();
    }
    if (i == 0) {
        float tc = red_c[0];
        out[0] = (tc > 0.0f) ? red_s[0] / (128.0f * tc) : 0.0f;
    }
}

extern "C" void kernel_launch(void* const* d_in, const int* in_sizes, int n_in,
                              void* d_out, int out_size, void* d_ws, size_t ws_size,
                              hipStream_t stream) {
    (void)in_sizes; (void)n_in; (void)out_size; (void)ws_size;
    const float* net_out   = (const float*)d_in[0];
    // d_in[1] = pt_offset (uniform segments; unused)
    const float* mask_embs = (const float*)d_in[2];
    const int*   mask_pts  = (const int*)d_in[3];
    // d_in[4] = logit_scale (unused by reference math)
    float* acc = (float*)d_ws;            // O*M*DE floats = 147,456 B

    const int accN = O_ * M_ * DE;
    zero_acc_kernel<<<(accN + 255) / 256, 256, 0, stream>>>(acc, accN);
    partial_kernel<<<O_ * (P_ / CHUNK), 256, 0, stream>>>(net_out, mask_pts, acc);
    finalize_kernel<<<1, 256, 0, stream>>>(acc, mask_embs, (float*)d_out);
}
